// LowRankConv2d_17222818857061
// MI455X (gfx1250) — matrix operand order repaired and checked by reading, not compiled, not run on hardware
//
#include <hip/hip_runtime.h>

typedef __attribute__((ext_vector_type(16))) _Float16 v16h;
typedef __attribute__((ext_vector_type(8)))  _Float16 v8h;
typedef __attribute__((ext_vector_type(8)))  float    v8f;
typedef __attribute__((ext_vector_type(4)))  unsigned int v4u;
typedef __attribute__((ext_vector_type(8)))  int      v8i;
typedef __attribute__((ext_vector_type(4)))  int      v4i;

#define O_CH   64
#define KK     17
#define MERGED_FLOATS (O_CH * 3 * KK * KK)          // 64*867 = 55488
#define BMAT_HALVES   (O_CH * KK * 32 * 16)         // 557056

// ---------------------------------------------------------------------------
// Kernel 1: build merged (64,3,17,17) fp32 weights. One block per output chan o.
// ---------------------------------------------------------------------------
__global__ __launch_bounds__(256)
void prep_weights(const float* __restrict__ A1, const float* __restrict__ B1,
                  const float* __restrict__ A2, const float* __restrict__ B2,
                  const float* __restrict__ A3, const float* __restrict__ B3,
                  const float* __restrict__ dw_w3, const float* __restrict__ dw_b3,
                  const float* __restrict__ w1_3,  const float* __restrict__ b1_3,
                  const float* __restrict__ w2_3,  const float* __restrict__ b2_3,
                  const float* __restrict__ dilw3,
                  const float* __restrict__ dw_w5, const float* __restrict__ dw_b5,
                  const float* __restrict__ w1_5,  const float* __restrict__ b1_5,
                  const float* __restrict__ w2_5,  const float* __restrict__ b2_5,
                  const float* __restrict__ dilw5,
                  const float* __restrict__ dw_w9, const float* __restrict__ dw_b9,
                  const float* __restrict__ w1_9,  const float* __restrict__ b1_9,
                  const float* __restrict__ w2_9,  const float* __restrict__ b2_9,
                  const float* __restrict__ dilw9,
                  float* __restrict__ merged_out)
{
    const int o = blockIdx.x;          // 0..63
    const int tid = threadIdx.x;

    __shared__ float wpad[3][20][20];  // zero-padded weight0 (covers pads for s=4,2,1)
    __shared__ float bufA[1200];       // depthwise output       (max 48*25)
    __shared__ float bufB[3888];       // pw1+relu output        (max 48*81)
    __shared__ float bufC[867];        // pw2 output             (max 3*289)
    __shared__ float bufD[243];        // bilinear output        (max 3*81)
    __shared__ float macc[867];        // merged accumulator (3,17,17)

    for (int i = tid; i < 1200; i += 256) ((float*)wpad)[i] = 0.f;
    for (int i = tid; i < 867;  i += 256) macc[i] = 0.f;
    __syncthreads();

    // weight0 = concat_p einsum('kj,jl->kl', A_p[o], B_p[o])
    const float* As[3] = {A1, A2, A3};
    const float* Bs[3] = {B1, B2, B3};
    for (int i = tid; i < 867; i += 256) {
        int p = i / 289, rem = i % 289, ky = rem / 17, kx = rem % 17;
        const float* Ap = As[p] + o * 68;   // (17,4)
        const float* Bp = Bs[p] + o * 68;   // (4,17)
        float s = 0.f;
        for (int r = 0; r < 4; ++r) s += Ap[ky * 4 + r] * Bp[r * 17 + kx];
        wpad[p][ky][kx] = s;
    }
    __syncthreads();

    const float* dww[3] = {dw_w3, dw_w5, dw_w9};
    const float* dwb[3] = {dw_b3, dw_b5, dw_b9};
    const float* pw1[3] = {w1_3, w1_5, w1_9};
    const float* pb1[3] = {b1_3, b1_5, b1_9};
    const float* pw2[3] = {w2_3, w2_5, w2_9};
    const float* pb2[3] = {b2_3, b2_5, b2_9};
    const float* dlw[3] = {dilw3, dilw5, dilw9};
    const int cN[3]   = {48, 12, 3};
    const int hidN[3] = {64, 48, 12};
    const int outN[3] = {12, 6, 3};
    const int spN[3]  = {5, 9, 17};
    const int scN[3]  = {4, 2, 1};
    const int ksN[3]  = {3, 5, 9};
    const int ndN[3]  = {3, 1, 1};
    const int dilsAll[3][3] = {{4, 5, 7}, {1, 0, 0}, {2, 0, 0}};

    for (int bi = 0; bi < 3; ++bi) {
        const int c = cN[bi], hid = hidN[bi], outc = outN[bi];
        const int sp = spN[bi], sc = scN[bi], ks = ksN[bi];

        // depthwise 3x3 (pad 1) on pixel-unshuffled weight0
        for (int i = tid; i < c * sp * sp; i += 256) {
            int ci = i / (sp * sp), rem = i % (sp * sp), y = rem / sp, xx = rem % sp;
            int p = ci / (sc * sc), s2 = ci % (sc * sc), sy = s2 / sc, sx = s2 % sc;
            float acc = dwb[bi][ci];
            for (int dy = 0; dy < 3; ++dy)
                for (int dx = 0; dx < 3; ++dx) {
                    int yy = y + dy - 1, xw = xx + dx - 1;
                    if (yy < 0 || yy >= sp || xw < 0 || xw >= sp) continue;
                    acc += wpad[p][yy * sc + sy][xw * sc + sx] * dww[bi][ci * 9 + dy * 3 + dx];
                }
            bufA[i] = acc;
        }
        __syncthreads();
        // 1x1 conv + relu
        for (int i = tid; i < hid * sp * sp; i += 256) {
            int hh = i / (sp * sp), rem = i % (sp * sp);
            float acc = pb1[bi][hh];
            for (int ci = 0; ci < c; ++ci) acc += bufA[ci * sp * sp + rem] * pw1[bi][hh * c + ci];
            bufB[i] = acc > 0.f ? acc : 0.f;
        }
        __syncthreads();
        // 1x1 conv
        for (int i = tid; i < outc * sp * sp; i += 256) {
            int oc2 = i / (sp * sp), rem = i % (sp * sp);
            float acc = pb2[bi][oc2];
            for (int hh = 0; hh < hid; ++hh) acc += bufB[hh * sp * sp + rem] * pw2[bi][oc2 * hid + hh];
            bufC[i] = acc;
        }
        __syncthreads();
        // bilinear sp -> ks
        for (int i = tid; i < outc * ks * ks; i += 256) {
            int oc2 = i / (ks * ks), rem = i % (ks * ks), oy = rem / ks, ox = rem % ks;
            float sy_ = fminf(fmaxf((oy + 0.5f) * ((float)sp / ks) - 0.5f, 0.f), sp - 1.f);
            float sx_ = fminf(fmaxf((ox + 0.5f) * ((float)sp / ks) - 0.5f, 0.f), sp - 1.f);
            int y0i = (int)floorf(sy_); int y1i = min(y0i + 1, sp - 1); float ty = sy_ - y0i;
            int x0i = (int)floorf(sx_); int x1i = min(x0i + 1, sp - 1); float tx = sx_ - x0i;
            const float* C = bufC + oc2 * sp * sp;
            float top = C[y0i * sp + x0i] * (1.f - tx) + C[y0i * sp + x1i] * tx;
            float bot = C[y1i * sp + x0i] * (1.f - tx) + C[y1i * sp + x1i] * tx;
            bufD[i] = top * (1.f - ty) + bot * ty;
        }
        __syncthreads();
        // dilation expansion + merge
        for (int id = 0; id < ndN[bi]; ++id) {
            int d = dilsAll[bi][id];
            int e = d * (ks - 1) + 1, pd = 8 - e / 2;
            for (int i = tid; i < 3 * ks * ks; i += 256) {
                int p = i / (ks * ks), rem = i % (ks * ks), h = rem / ks, wd = rem % ks;
                float acc = 0.f;
                for (int cc = 0; cc < outc; ++cc)
                    acc += bufD[cc * ks * ks + rem] * dlw[bi][(id * 3 + p) * outc + cc];
                macc[p * 289 + (pd + h * d) * 17 + (pd + wd * d)] += acc;
            }
            __syncthreads();
        }
    }
    for (int i = tid; i < 867; i += 256) merged_out[o * 867 + i] = macc[i];
}

// ---------------------------------------------------------------------------
// Kernel 2: pack B fragments in WMMA lane order, fp32 -> f16.
// Per (ch, r): 32 lanes x 16 halves. lane L holds N=L%16, K = (L<16 ? e : 16+e).
// Column n = 3*delta + oc (n<15): B[c][n] = w[oc, r, c - delta] for c-delta in [0,16].
// ---------------------------------------------------------------------------
__global__ __launch_bounds__(256)
void prep_bmat(const float* __restrict__ merged, _Float16* __restrict__ bmat)
{
    int idx = blockIdx.x * 256 + threadIdx.x;
    if (idx >= BMAT_HALVES) return;
    int ch  = idx / (KK * 512);
    int rem = idx % (KK * 512);
    int r   = rem / 512;
    int q   = rem % 512;
    int lane = q / 16, e = q % 16;
    int n = lane & 15;
    int ccol = (lane < 16) ? e : (16 + e);
    float v = 0.f;
    if (n < 15) {
        int dlt = n / 3, oc = n % 3;
        int j = ccol - dlt;
        if (j >= 0 && j <= 16)
            v = merged[ch * 867 + oc * 289 + r * 17 + j];
    }
    bmat[idx] = (_Float16)v;
}

// ---------------------------------------------------------------------------
// Kernel 3: main 17x17 grouped conv via WMMA f32_16x16x32_f16.
// Block = 128 threads (4 waves). Each wave: one x-anchor -> 16y x 5x x 3oc tile.
// ---------------------------------------------------------------------------
__global__ __launch_bounds__(128)
void dkconv_main(const float* __restrict__ x, const _Float16* __restrict__ bmat,
                 float* __restrict__ out)
{
    __shared__ v8h   ldsBv[KK * 64];        // 17*512 halves = 17408 B, WMMA B frags
    __shared__ float win[32 * 48];          // fp32 input window (y0-8.., x0-8..)
    __shared__ v8h   reg[4][32][4];         // 4 per-wave aligned 32x32 f16 regions

    const int tid  = threadIdx.x;
    const int wave = tid >> 5;
    const int lane = tid & 31;

    const int bc = blockIdx.z;
    const int b  = bc >> 6;
    const int ch = bc & 63;
    const int y0 = blockIdx.y << 4;
    const int x0 = blockIdx.x * 20;

    // ---- bring this channel's B fragments (17408 B) into LDS -------------
    const _Float16* bsrc = bmat + (size_t)ch * (KK * 512);
#if defined(__gfx1250__) && __has_builtin(__builtin_amdgcn_tensor_load_to_lds)
    if (tid < 32) {   // single wave issues the TDM op (EXEC ignored by TDM)
        unsigned long long ga = (unsigned long long)(uintptr_t)bsrc;
        unsigned lds_off = (unsigned)(uintptr_t)(&ldsBv[0]);
        v4u g0;
        g0[0] = 1u;                                   // count=1 valid, no gather
        g0[1] = lds_off;                              // lds_addr
        g0[2] = (unsigned)ga;                         // global_addr lo
        g0[3] = (unsigned)(ga >> 32) | 0x80000000u;   // global_addr hi | type=2
        v8i g1;
        g1[0] = (int)(2u << 16);                      // data_size = 4 bytes
        g1[1] = (int)((4352u & 0xFFFFu) << 16);       // tensor_dim0 lo (4352 dwords)
        g1[2] = (int)((4352u >> 16) | (1u << 16));    // tensor_dim0 hi | tensor_dim1 lo
        g1[3] = (int)(4352u << 16);                   // tensor_dim1 hi | tile_dim0
        g1[4] = 1;                                    // tile_dim1 = 1
        g1[5] = 4352;                                 // tensor_dim0_stride lo
        g1[6] = 0; g1[7] = 0;
        v4i z4 = {0, 0, 0, 0};
#if __clang_major__ >= 23
        v8i z8 = {0, 0, 0, 0, 0, 0, 0, 0};
        __builtin_amdgcn_tensor_load_to_lds(g0, g1, z4, z4, z8, 0);
#else
        __builtin_amdgcn_tensor_load_to_lds(g0, g1, z4, z4, 0);
#endif
        __builtin_amdgcn_s_wait_tensorcnt(0);
    }
#else
    for (int i = tid; i < KK * 512 / 2; i += 128)
        ((unsigned*)ldsBv)[i] = ((const unsigned*)bsrc)[i];
#endif

    // ---- stage fp32 input window 32y x 48x (zero padded) ------------------
    const float* xch = x + (size_t)(b * 64 + ch) * 256 * 256;
    for (int i = tid; i < 32 * 48; i += 128) {
        int row = i / 48, col = i % 48;
        int gy = y0 - 8 + row, gx = x0 - 8 + col;
        float v = 0.f;
        if (gy >= 0 && gy < 256 && gx >= 0 && gx < 256) v = xch[gy * 256 + gx];
        win[i] = v;
    }
    __syncthreads();

    // ---- build per-wave 16B-aligned f16 regions (region w: window cols 5w..5w+31)
    _Float16* regh = (_Float16*)reg;
    for (int w = 0; w < 4; ++w)
        for (int i = tid; i < 1024; i += 128) {
            int row = i >> 5, col = i & 31;
            regh[w * 1024 + i] = (_Float16)win[row * 48 + 5 * w + col];
        }
    __syncthreads();

    // ---- 17 chained WMMAs: D[m, 3d+oc] = out[ch*3+oc, y0+m, x0+5*wave+d] ----
    v8f acc = {};
    const int m  = lane & 15;
    const int hi = lane >> 4;
    for (int r = 0; r < KK; ++r) {
        v8h a0 = reg[wave][m + r][2 * hi];        // K runs {0..7} / {16..23}
        v8h a1 = reg[wave][m + r][2 * hi + 1];    // K runs {8..15} / {24..31}
        v8h b0 = ldsBv[(r * 32 + lane) * 2];
        v8h b1 = ldsBv[(r * 32 + lane) * 2 + 1];
        union { v16h v; v8h h[2]; } ua, ub;
        ua.h[0] = a0; ua.h[1] = a1;
        ub.h[0] = b0; ub.h[1] = b1;
        acc = __builtin_amdgcn_wmma_f32_16x16x32_f16(
            false, ua.v, false, ub.v, (short)0, acc, false, false);
    }

    // ---- scatter results: lane n<15 -> (delta, oc); element e -> m = hi*8+e
    const int n = lane & 15;
    if (n < 15) {
        int dlt = n / 3, oc = n % 3;
        int xo = x0 + wave * 5 + dlt;
        if (xo < 256) {
            float* op = out + (((size_t)b * 192 + (ch * 3 + oc)) * 256) * 256 + xo;
            int mb = hi * 8;
#pragma unroll
            for (int e = 0; e < 8; ++e)
                op[(size_t)(y0 + mb + e) * 256] = acc[e];
        }
    }
}

// ---------------------------------------------------------------------------
extern "C" void kernel_launch(void* const* d_in, const int* in_sizes, int n_in,
                              void* d_out, int out_size, void* d_ws, size_t ws_size,
                              hipStream_t stream)
{
    // setup_inputs() dict order: x, A1, B1, A2, B2, A3, B3,
    //   then for k in (3,5,9): dw_w, dw_b, w1, b1, w2, b2, dilw
    const float* x   = (const float*)d_in[0];
    const float* A1  = (const float*)d_in[1];
    const float* B1  = (const float*)d_in[2];
    const float* A2  = (const float*)d_in[3];
    const float* B2  = (const float*)d_in[4];
    const float* A3  = (const float*)d_in[5];
    const float* B3  = (const float*)d_in[6];
    const float* dw_w3 = (const float*)d_in[7];
    const float* dw_b3 = (const float*)d_in[8];
    const float* w1_3  = (const float*)d_in[9];
    const float* b1_3  = (const float*)d_in[10];
    const float* w2_3  = (const float*)d_in[11];
    const float* b2_3  = (const float*)d_in[12];
    const float* dilw3 = (const float*)d_in[13];
    const float* dw_w5 = (const float*)d_in[14];
    const float* dw_b5 = (const float*)d_in[15];
    const float* w1_5  = (const float*)d_in[16];
    const float* b1_5  = (const float*)d_in[17];
    const float* w2_5  = (const float*)d_in[18];
    const float* b2_5  = (const float*)d_in[19];
    const float* dilw5 = (const float*)d_in[20];
    const float* dw_w9 = (const float*)d_in[21];
    const float* dw_b9 = (const float*)d_in[22];
    const float* w1_9  = (const float*)d_in[23];
    const float* b1_9  = (const float*)d_in[24];
    const float* w2_9  = (const float*)d_in[25];
    const float* b2_9  = (const float*)d_in[26];
    const float* dilw9 = (const float*)d_in[27];

    float*    merged = (float*)d_ws;                               // 221,952 B
    _Float16* bmat   = (_Float16*)((char*)d_ws + 224 * 1024);      // 1,114,112 B

    prep_weights<<<O_CH, 256, 0, stream>>>(
        A1, B1, A2, B2, A3, B3,
        dw_w3, dw_b3, w1_3, b1_3, w2_3, b2_3, dilw3,
        dw_w5, dw_b5, w1_5, b1_5, w2_5, b2_5, dilw5,
        dw_w9, dw_b9, w1_9, b1_9, w2_9, b2_9, dilw9,
        merged);

    prep_bmat<<<(BMAT_HALVES + 255) / 256, 256, 0, stream>>>(merged, bmat);

    // grid: 13 x-slabs (20 out cols each), 16 y-tiles, 8 batch * 64 channels
    dkconv_main<<<dim3(13, 16, 8 * 64), 128, 0, stream>>>(x, bmat, (float*)d_out);
}
